// VariationalTransformerEncoder_54752243090109
// MI455X (gfx1250) — matrix operand order repaired and checked by reading, not compiled, not run on hardware
//
#include <hip/hip_runtime.h>

// ---------------------------------------------------------------------------
// Types for CDNA5 WMMA (wave32): V_WMMA_F32_16X16X32_BF16
// ---------------------------------------------------------------------------
typedef __attribute__((ext_vector_type(16))) __bf16 bf16x16;
typedef __attribute__((ext_vector_type(8)))  float  floatx8;

union FragBF {
    bf16x16 v;
    uint4   q[2];   // 16 x bf16 = 32 bytes
};

__device__ __forceinline__ floatx8 bc8(float x) {
    floatx8 v = {x, x, x, x, x, x, x, x};
    return v;
}

// ---------------------------------------------------------------------------
// Helpers
// ---------------------------------------------------------------------------
__device__ __forceinline__ unsigned short f2bf(float f) {
    unsigned u = __float_as_uint(f);
    unsigned r = (u + 0x7FFFu + ((u >> 16) & 1u)) >> 16;   // round-to-nearest-even
    return (unsigned short)r;
}
__device__ __forceinline__ float bf2f(unsigned short u) {
    return __uint_as_float(((unsigned)u) << 16);
}
// Order-preserving map: float -> uint for atomicMax on signed floats
__device__ __forceinline__ unsigned enc_ord(float f) {
    unsigned b = __float_as_uint(f);
    return (b & 0x80000000u) ? ~b : (b | 0x80000000u);
}
__device__ __forceinline__ float dec_ord(unsigned u) {
    return (u & 0x80000000u) ? __uint_as_float(u ^ 0x80000000u)
                             : __uint_as_float(~u);
}
__device__ __forceinline__ void atomAddF(float* p, float v) {
    __hip_atomic_fetch_add(p, v, __ATOMIC_RELAXED, __HIP_MEMORY_SCOPE_AGENT);
}

// ---------------------------------------------------------------------------
// fp32 -> bf16 bulk convert
// ---------------------------------------------------------------------------
__global__ void k_f32_to_bf16(const float* __restrict__ in,
                              unsigned short* __restrict__ out, long n) {
    long t = (long)blockIdx.x * blockDim.x + threadIdx.x;
    if (t < n) out[t] = f2bf(in[t]);
}

// Transpose fp32 weight [K,D] row-major -> bf16 [D,K] row-major
__global__ void k_transpose_bf16(const float* __restrict__ W,
                                 unsigned short* __restrict__ Wt,
                                 int K, int D) {
    long t = (long)blockIdx.x * blockDim.x + threadIdx.x;
    long n = (long)K * D;
    if (t >= n) return;
    int kk = (int)(t / D);
    int nn = (int)(t - (long)kk * D);
    Wt[(size_t)nn * K + kk] = f2bf(W[t]);
}

// ---------------------------------------------------------------------------
// Fused 4-output WMMA GEMM: {q,k,v,s}[M,D] = A[M,K](bf16) x Wt_j[D,K](bf16)
// + bias_j.  One wave per 16x16 tile position; A-fragment loaded once per
// K-step and reused by 4 WMMA chains. All 5 fragment loads are issued before
// the 4 WMMAs so the loads clause together and WMMAs retire under partial
// s_wait_loadcnt. M%16==0, K%32==0, D%16==0.
// A-frag (16x32): lanes 0-15 K 0..15, lanes 16-31 K
// 16..31 (same as B). B-frag (32x16): lanes 0-15 K 0..15, lanes 16-31 K 16..31.
// ---------------------------------------------------------------------------
__global__ void k_qkvs_wmma(const unsigned short* __restrict__ A,
                            const unsigned short* __restrict__ Btq,
                            const unsigned short* __restrict__ Btk,
                            const unsigned short* __restrict__ Btv,
                            const unsigned short* __restrict__ Bts,
                            const float* __restrict__ bq,
                            const float* __restrict__ bk,
                            const float* __restrict__ bv,
                            const float* __restrict__ bs,
                            float* __restrict__ Cq, float* __restrict__ Ck,
                            float* __restrict__ Cv, float* __restrict__ Cs,
                            int M, int K, int D) {
    const int wid    = (int)((blockIdx.x * blockDim.x + threadIdx.x) >> 5);
    const int tilesN = D >> 4;
    const int total  = (M >> 4) * tilesN;
    if (wid >= total) return;
    const int tm   = wid / tilesN;
    const int tn   = wid - tm * tilesN;
    const int lane = threadIdx.x & 31;
    const int hl   = lane >> 4;
    const int l16  = lane & 15;

    const size_t abase = (size_t)((tm << 4) + l16) * K;
    const size_t bbase = (size_t)((tn << 4) + l16) * K;
    const int aoff = hl << 3;    // 0 or 8
    const int boff = hl << 4;    // 0 or 16
    const int colb = (tn << 4) + l16;

    floatx8 aq = bc8(bq[colb]);
    floatx8 ak = bc8(bk[colb]);
    floatx8 av = bc8(bv[colb]);
    floatx8 as = bc8(bs[colb]);

    for (int kb = 0; kb < K; kb += 32) {
        if (kb + 32 < K) __builtin_prefetch(A + abase + kb + 32, 0, 0);
        FragBF a, bQ, bK, bV, bS;
        // Issue all loads first -> one VMEM clause, staggered waits.
        a.q[0]  = *(const uint4*)(A   + abase + kb + boff);
        a.q[1]  = *(const uint4*)(A   + abase + kb + boff + 8);
        bQ.q[0] = *(const uint4*)(Btq + bbase + kb + boff);
        bQ.q[1] = *(const uint4*)(Btq + bbase + kb + boff + 8);
        bK.q[0] = *(const uint4*)(Btk + bbase + kb + boff);
        bK.q[1] = *(const uint4*)(Btk + bbase + kb + boff + 8);
        bV.q[0] = *(const uint4*)(Btv + bbase + kb + boff);
        bV.q[1] = *(const uint4*)(Btv + bbase + kb + boff + 8);
        bS.q[0] = *(const uint4*)(Bts + bbase + kb + boff);
        bS.q[1] = *(const uint4*)(Bts + bbase + kb + boff + 8);

        aq = __builtin_amdgcn_wmma_f32_16x16x32_bf16(false, a.v, false, bQ.v, (short)0, aq, false, false);
        ak = __builtin_amdgcn_wmma_f32_16x16x32_bf16(false, a.v, false, bK.v, (short)0, ak, false, false);
        av = __builtin_amdgcn_wmma_f32_16x16x32_bf16(false, a.v, false, bV.v, (short)0, av, false, false);
        as = __builtin_amdgcn_wmma_f32_16x16x32_bf16(false, a.v, false, bS.v, (short)0, as, false, false);
    }

#pragma unroll
    for (int r = 0; r < 8; ++r) {
        const int rowC = (tm << 4) + r + (hl << 3);
        const size_t o = (size_t)rowC * D + colb;
        Cq[o] = aq[r];
        Ck[o] = ak[r];
        Cv[o] = av[r];
        Cs[o] = as[r];
    }
}

// ---------------------------------------------------------------------------
// Attention passes
// ---------------------------------------------------------------------------
__global__ void k_init(float* __restrict__ att, unsigned* __restrict__ m_u,
                       float* __restrict__ ssum, int N, int d) {
    long t  = (long)blockIdx.x * blockDim.x + threadIdx.x;
    long nd = (long)N * d;
    if (t < nd) att[t] = 0.0f;
    if (t < N) { m_u[t] = 0u; ssum[t] = 0.0f; }
}

// Pass 1 (WMMA-fused): one wave owns 16 edges. eproj tile [16 edges x d] is
// computed in-register via WMMA from edge_attr (bf16, K=64) and the
// transposed edge weights; C layout (row = r + 8*half, col = lane%16) feeds
// per-edge gathers of q[dst]/k[src] at matching feature columns. Dot is
// reduced across the 16 lanes of each half with xor-shuffles.
__global__ void k_edge_logits_wmma(const float* __restrict__ qf,
                                   const float* __restrict__ kfm,
                                   const unsigned short* __restrict__ eab,
                                   const unsigned short* __restrict__ ewT,
                                   const float* __restrict__ ebias,
                                   const int* __restrict__ src,
                                   const int* __restrict__ dst,
                                   float* __restrict__ logits,
                                   unsigned* __restrict__ m_u,
                                   int E, int d, float scale) {
    const int wid    = (int)((blockIdx.x * blockDim.x + threadIdx.x) >> 5);
    const int ntile  = E >> 4;
    if (wid >= ntile) return;
    const int lane = threadIdx.x & 31;
    const int hl   = lane >> 4;
    const int l16  = lane & 15;
    const int base = wid << 4;

    const unsigned short* arow = eab + (size_t)(base + l16) * 64;
    const int aoff = hl << 3;
    const int boff = hl << 4;

    FragBF a0, a1;
    a0.q[0] = *(const uint4*)(arow + boff);
    a0.q[1] = *(const uint4*)(arow + boff + 8);
    a1.q[0] = *(const uint4*)(arow + 32 + boff);
    a1.q[1] = *(const uint4*)(arow + 32 + boff + 8);

    int sr[8], dr[8];
#pragma unroll
    for (int r = 0; r < 8; ++r) {
        const int er = base + r + (hl << 3);
        sr[r] = src[er];
        dr[r] = dst[er];
    }

    float preg[8] = {0.f, 0.f, 0.f, 0.f, 0.f, 0.f, 0.f, 0.f};
    const int ct = d >> 4;
    for (int c = 0; c < ct; ++c) {
        const unsigned short* brow = ewT + (size_t)((c << 4) + l16) * 64;
        FragBF b0, b1;
        b0.q[0] = *(const uint4*)(brow + boff);
        b0.q[1] = *(const uint4*)(brow + boff + 8);
        b1.q[0] = *(const uint4*)(brow + 32 + boff);
        b1.q[1] = *(const uint4*)(brow + 32 + boff + 8);

        floatx8 acc = bc8(ebias[(c << 4) + l16]);
        acc = __builtin_amdgcn_wmma_f32_16x16x32_bf16(false, a0.v, false, b0.v, (short)0, acc, false, false);
        acc = __builtin_amdgcn_wmma_f32_16x16x32_bf16(false, a1.v, false, b1.v, (short)0, acc, false, false);

        const int feat = (c << 4) + l16;
#pragma unroll
        for (int r = 0; r < 8; ++r) {
            const float qv = qf [(size_t)dr[r] * d + feat];
            const float kv = kfm[(size_t)sr[r] * d + feat];
            preg[r] += qv * (kv + acc[r]);
        }
    }

#pragma unroll
    for (int r = 0; r < 8; ++r) {
        float p = preg[r];
        p += __shfl_xor(p, 1, 32);
        p += __shfl_xor(p, 2, 32);
        p += __shfl_xor(p, 4, 32);
        p += __shfl_xor(p, 8, 32);
        if (l16 == 0) {
            const int er = base + r + (hl << 3);
            p *= scale;
            logits[er] = p;
            atomicMax(m_u + dr[r], enc_ord(p));
        }
    }
}

// Pass 2: ex = exp(logit - max[dst]); atomicAdd sum[dst]. Thread per edge.
__global__ void k_edge_exp(float* __restrict__ logits,
                           const unsigned* __restrict__ m_u,
                           const int* __restrict__ dst,
                           float* __restrict__ ssum, int E) {
    int t = blockIdx.x * blockDim.x + threadIdx.x;
    if (t >= E) return;
    int dd = dst[t];
    float ex = __expf(logits[t] - dec_ord(m_u[dd]));
    logits[t] = ex;
    atomAddF(ssum + dd, ex);
}

// Pass 3 (WMMA-fused): recompute eproj tile in-register, then
// att[dst] += (v[src] + eproj) * alpha per element via fp32 atomics.
__global__ void k_edge_accum_wmma(const float* __restrict__ ex,
                                  const float* __restrict__ ssum,
                                  const float* __restrict__ vf,
                                  const unsigned short* __restrict__ eab,
                                  const unsigned short* __restrict__ ewT,
                                  const float* __restrict__ ebias,
                                  const int* __restrict__ src,
                                  const int* __restrict__ dst,
                                  float* __restrict__ att,
                                  int E, int d) {
    const int wid   = (int)((blockIdx.x * blockDim.x + threadIdx.x) >> 5);
    const int ntile = E >> 4;
    if (wid >= ntile) return;
    const int lane = threadIdx.x & 31;
    const int hl   = lane >> 4;
    const int l16  = lane & 15;
    const int base = wid << 4;

    const unsigned short* arow = eab + (size_t)(base + l16) * 64;
    const int aoff = hl << 3;
    const int boff = hl << 4;

    FragBF a0, a1;
    a0.q[0] = *(const uint4*)(arow + boff);
    a0.q[1] = *(const uint4*)(arow + boff + 8);
    a1.q[0] = *(const uint4*)(arow + 32 + boff);
    a1.q[1] = *(const uint4*)(arow + 32 + boff + 8);

    int sr[8], dr[8];
    float al[8];
#pragma unroll
    for (int r = 0; r < 8; ++r) {
        const int er = base + r + (hl << 3);
        sr[r] = src[er];
        dr[r] = dst[er];
        al[r] = ex[er] / (ssum[dr[r]] + 1e-16f);
    }

    const int ct = d >> 4;
    for (int c = 0; c < ct; ++c) {
        const unsigned short* brow = ewT + (size_t)((c << 4) + l16) * 64;
        FragBF b0, b1;
        b0.q[0] = *(const uint4*)(brow + boff);
        b0.q[1] = *(const uint4*)(brow + boff + 8);
        b1.q[0] = *(const uint4*)(brow + 32 + boff);
        b1.q[1] = *(const uint4*)(brow + 32 + boff + 8);

        floatx8 acc = bc8(ebias[(c << 4) + l16]);
        acc = __builtin_amdgcn_wmma_f32_16x16x32_bf16(false, a0.v, false, b0.v, (short)0, acc, false, false);
        acc = __builtin_amdgcn_wmma_f32_16x16x32_bf16(false, a1.v, false, b1.v, (short)0, acc, false, false);

        const int feat = (c << 4) + l16;
#pragma unroll
        for (int r = 0; r < 8; ++r) {
            const float val = (vf[(size_t)sr[r] * d + feat] + acc[r]) * al[r];
            atomAddF(att + (size_t)dr[r] * d + feat, val);
        }
    }
}

// Layer-1 epilogue: h = relu(att + skip); emit bf16 activations for next GEMMs.
__global__ void k_relu_bf(const float* __restrict__ att,
                          const float* __restrict__ slin,
                          unsigned short* __restrict__ hb, long n) {
    long t = (long)blockIdx.x * blockDim.x + threadIdx.x;
    if (t >= n) return;
    float x = att[t] + slin[t];
    hb[t] = f2bf(x > 0.0f ? x : 0.0f);
}

// mu/ls epilogue: out = att + skip (fp32, straight into d_out).
__global__ void k_add_out(const float* __restrict__ att,
                          const float* __restrict__ slin,
                          float* __restrict__ out, long n) {
    long t = (long)blockIdx.x * blockDim.x + threadIdx.x;
    if (t < n) out[t] = att[t] + slin[t];
}

// ---------------------------------------------------------------------------
// Host orchestration
// ---------------------------------------------------------------------------
extern "C" void kernel_launch(void* const* d_in, const int* in_sizes, int n_in,
                              void* d_out, int out_size, void* d_ws, size_t ws_size,
                              hipStream_t stream) {
    (void)n_in; (void)out_size; (void)ws_size;
    const float* x  = (const float*)d_in[0];
    const int*   ei = (const int*)d_in[1];
    const float* ea = (const float*)d_in[2];
    const int N = in_sizes[0] / 256;
    const int E = in_sizes[2] / 64;
    const int* src = ei;
    const int* dst = ei + E;

    // Weight/bias pointers: layer*5 + {0:q,1:k,2:v,3:e,4:s}
    const float* W[15]; const float* B[15];
    for (int i = 0; i < 15; ++i) {
        W[i] = (const float*)d_in[3 + 2 * i];
        B[i] = (const float*)d_in[4 + 2 * i];
    }

    // Workspace carve (256B aligned)
    char* base = (char*)d_ws;
    size_t off = 0;
    auto carve = [&](size_t bytes) -> void* {
        void* r = base + off;
        off += (bytes + 255) & ~(size_t)255;
        return r;
    };
    unsigned short* xb  = (unsigned short*)carve((size_t)N * 256 * 2);
    unsigned short* eab = (unsigned short*)carve((size_t)E * 64 * 2);
    unsigned short* wt[5];
    for (int j = 0; j < 5; ++j) wt[j] = (unsigned short*)carve(32768 * 2);
    float* qf     = (float*)carve((size_t)N * 128 * 4);
    float* kf     = (float*)carve((size_t)N * 128 * 4);
    float* vf     = (float*)carve((size_t)N * 128 * 4);
    float* slin   = (float*)carve((size_t)N * 128 * 4);
    float* logits = (float*)carve((size_t)E * 4);
    unsigned* m_u = (unsigned*)carve((size_t)N * 4);
    float* ssum   = (float*)carve((size_t)N * 4);
    float* att    = (float*)carve((size_t)N * 128 * 4);
    unsigned short* hb = (unsigned short*)carve((size_t)N * 128 * 2);

    const int TB = 256;
    auto blocks1 = [&](long n) { return (int)((n + TB - 1) / TB); };

    // One-time input conversions
    {
        long nx = (long)N * 256;
        k_f32_to_bf16<<<blocks1(nx), TB, 0, stream>>>(x, xb, nx);
        long ne = (long)E * 64;
        k_f32_to_bf16<<<blocks1(ne), TB, 0, stream>>>(ea, eab, ne);
    }

    auto run_layer = [&](const unsigned short* act, int Kact, int d,
                         int wbase, float scale) {
        // Pre-transpose weights to bf16 [D,K]
        auto prep = [&](int widx, int Kd, int Dd, int slot) {
            long n = (long)Kd * Dd;
            k_transpose_bf16<<<blocks1(n), TB, 0, stream>>>(W[widx], wt[slot], Kd, Dd);
        };
        prep(wbase + 0, Kact, d, 0);   // q
        prep(wbase + 1, Kact, d, 1);   // k
        prep(wbase + 2, Kact, d, 2);   // v
        prep(wbase + 3, 64,   d, 3);   // e
        prep(wbase + 4, Kact, d, 4);   // s

        // Fused q/k/v/skip projection
        {
            long waves = (long)(N / 16) * (d / 16);
            int blk = (int)((waves * 32 + TB - 1) / TB);
            k_qkvs_wmma<<<blk, TB, 0, stream>>>(
                act, wt[0], wt[1], wt[2], wt[4],
                B[wbase + 0], B[wbase + 1], B[wbase + 2], B[wbase + 4],
                qf, kf, vf, slin, N, Kact, d);
        }

        long nd = (long)N * d;
        k_init<<<blocks1(nd), TB, 0, stream>>>(att, m_u, ssum, N, d);

        // Edge passes: 1 wave per 16 edges, eproj recomputed in-register
        long ewaves = (long)(E / 16);
        int eb = (int)((ewaves * 32 + TB - 1) / TB);
        k_edge_logits_wmma<<<eb, TB, 0, stream>>>(
            qf, kf, eab, wt[3], B[wbase + 3], src, dst, logits, m_u, E, d, scale);
        k_edge_exp<<<blocks1(E), TB, 0, stream>>>(logits, m_u, dst, ssum, E);
        k_edge_accum_wmma<<<eb, TB, 0, stream>>>(
            logits, ssum, vf, eab, wt[3], B[wbase + 3], src, dst, att, E, d);
    };

    // Layer c1: IN=256 -> H=128, relu
    run_layer(xb, 256, 128, 0, 0.08838834764831845f);   // 1/sqrt(128)
    {
        long nh = (long)N * 128;
        k_relu_bf<<<blocks1(nh), TB, 0, stream>>>(att, slin, hb, nh);
    }

    // Layer mu: H=128 -> OUT=64
    run_layer(hb, 128, 64, 5, 0.125f);                  // 1/sqrt(64)
    {
        long no = (long)N * 64;
        k_add_out<<<blocks1(no), TB, 0, stream>>>(att, slin, (float*)d_out, no);
    }

    // Layer ls: H=128 -> OUT=64
    run_layer(hb, 128, 64, 10, 0.125f);
    {
        long no = (long)N * 64;
        k_add_out<<<blocks1(no), TB, 0, stream>>>(att, slin, (float*)d_out + (size_t)N * 64, no);
    }
}
